// BiViewCompatibilityWeightedGATv2_28492813041840
// MI455X (gfx1250) — matrix operand order repaired and checked by reading, not compiled, not run on hardware
//
#include <hip/hip_runtime.h>
#include <hip/hip_bf16.h>
#include <stdint.h>

// ---------------------------------------------------------------------------
// BiView compatibility-weighted GATv2 for MI455X (gfx1250, wave32, WMMA).
//
// Roofline: GEMMs total ~6 GFLOP (tiny); edge gathers/scatters ~5 GB of HBM
// traffic dominate (~0.2 ms @ 23.3 TB/s). GEMMs use v_wmma_f32_16x16x32_f16
// (f16 A/B, f32 accum), fully unrolled via template K so the WMMA chain is
// static; A rows load as b128, B fragments come from transposed-LDS b128.
// ---------------------------------------------------------------------------

#define N_NODES     50000
#define N_EDGES     800000
#define NUM_GRAPHS  256
#define NUM_FEAT    128
#define NHID        64
#define NUM_CLASSES 10
#define NEG_SLOPE   0.2f

typedef __attribute__((ext_vector_type(16))) _Float16 v16h;
typedef __attribute__((ext_vector_type(8)))  _Float16 v8h;
typedef __attribute__((ext_vector_type(8)))  float    v8f;

static __device__ __forceinline__ float finite_or_zero(float x) {
  return (x >= -3.0e38f && x <= 3.0e38f) ? x : 0.0f;
}

// float atomic max via int/uint punning (works for mixed signs, -inf init)
static __device__ __forceinline__ void atomicMaxFloat(float* addr, float val) {
  if (val >= 0.0f) atomicMax((int*)addr, __float_as_int(val));
  else             atomicMin((unsigned int*)addr, __float_as_uint(val));
}

// ---------------------------------------------------------------------------
// WMMA GEMM: out[N x 64] = A[N x K] @ W[K x 64] (+ bias), K in {64,128}.
// 256 threads = 8 waves; wave computes a 16-row x 64-col tile -> 128 rows/blk.
// W is staged transposed into LDS as f16 so B fragments are contiguous b128.
// A fragments are preloaded once and reused across all four 16-col tiles.
// ---------------------------------------------------------------------------
template <int K>
__global__ void __launch_bounds__(256)
k_gemm64(const float* __restrict__ A, const float* __restrict__ W,
         const float* __restrict__ bias, float* __restrict__ out, int N) {
  constexpr int KT = K / 32;
  __shared__ _Float16 sWt[64 * K];    // transposed: sWt[n*K + k]
  __shared__ float    sB[64];

  const int tid = threadIdx.x;
  for (int idx = tid; idx < K * 64; idx += 256) {
    int k = idx >> 6, n = idx & 63;                 // W row-major [K,64]
    sWt[n * K + k] = (_Float16)W[idx];
  }
  if (tid < 64) sB[tid] = bias ? bias[tid] : 0.0f;
  __syncthreads();

  const int lane = tid & 31;
  const int wave = tid >> 5;
  const int rowBase = blockIdx.x * 128 + wave * 16;
  const int hl   = lane >> 4;          // 0: K-low half lanes, 1: K-high half
  const int col  = lane & 15;          // N within 16-wide tile / A row id
  int rA = rowBase + col;              // A row this lane feeds
  if (rA >= N) rA = N - 1;             // clamp: garbage rows land in unstored D rows
  const float4* Arow4 = (const float4*)(A + (size_t)rA * K);

  // Preload A fragments (16x32 f16 each): lanes 0-15 hold K=kt*32+{0..7,16..23},
  // lanes 16-31 hold K=kt*32+{8..15,24..31}; row = lane&15.
  v16h aF[KT];
  #pragma unroll
  for (int kt = 0; kt < KT; ++kt) {
    float4 lo0 = Arow4[kt * 8 + hl * 2 + 0];
    float4 lo1 = Arow4[kt * 8 + hl * 2 + 1];
    float4 hi0 = Arow4[kt * 8 + 4 + hl * 2 + 0];
    float4 hi1 = Arow4[kt * 8 + 4 + hl * 2 + 1];
    aF[kt][0]  = (_Float16)lo0.x; aF[kt][1]  = (_Float16)lo0.y;
    aF[kt][2]  = (_Float16)lo0.z; aF[kt][3]  = (_Float16)lo0.w;
    aF[kt][4]  = (_Float16)lo1.x; aF[kt][5]  = (_Float16)lo1.y;
    aF[kt][6]  = (_Float16)lo1.z; aF[kt][7]  = (_Float16)lo1.w;
    aF[kt][8]  = (_Float16)hi0.x; aF[kt][9]  = (_Float16)hi0.y;
    aF[kt][10] = (_Float16)hi0.z; aF[kt][11] = (_Float16)hi0.w;
    aF[kt][12] = (_Float16)hi1.x; aF[kt][13] = (_Float16)hi1.y;
    aF[kt][14] = (_Float16)hi1.z; aF[kt][15] = (_Float16)hi1.w;
  }

  const bool fullTile = (rowBase + 16 <= N);   // wave-uniform fast path
  #pragma unroll
  for (int nt = 0; nt < 4; ++nt) {
    v8f acc = {};
    const int n = nt * 16 + col;
    #pragma unroll
    for (int kt = 0; kt < KT; ++kt) {
      // B fragment (32x16 f16): lane holds K=kb+{0..7,16..23}, col n.
      const int kb = kt * 32 + hl * 8;
      v8h b0 = *(const v8h*)&sWt[n * K + kb];
      v8h b1 = *(const v8h*)&sWt[n * K + kb + 16];
      v16h b = __builtin_shufflevector(b0, b1, 0, 1, 2, 3, 4, 5, 6, 7,
                                                8, 9, 10, 11, 12, 13, 14, 15);
      acc = __builtin_amdgcn_wmma_f32_16x16x32_f16(
          /*neg_a=*/false, aF[kt], /*neg_b=*/false, b,
          /*c_mod=*/(short)0, acc, /*reuse_a=*/false, /*reuse_b=*/false);
    }
    // D layout: VGPR r -> M = (lane>>4)*8 + r, N = lane&15
    const int oRow0 = rowBase + hl * 8;
    const float bv  = sB[n];
    if (fullTile) {
      #pragma unroll
      for (int r = 0; r < 8; ++r)
        out[(size_t)(oRow0 + r) * 64 + n] = acc[r] + bv;
    } else {
      #pragma unroll
      for (int r = 0; r < 8; ++r) {
        int row = oRow0 + r;
        if (row < N) out[(size_t)row * 64 + n] = acc[r] + bv;
      }
    }
  }
}

// ---------------------------------------------------------------------------
// Init kernels
// ---------------------------------------------------------------------------
__global__ void k_init_conv(float* m, float* den, float* agg, int N) {
  int i = blockIdx.x * blockDim.x + threadIdx.x;
  if (i < N * 64) agg[i] = 0.0f;
  if (i < N) { m[i] = -__builtin_inff(); den[i] = 0.0f; }
}

__global__ void k_init_graphseg(float* mx, float* sm) {
  int i = blockIdx.x * blockDim.x + threadIdx.x;
  if (i < NUM_GRAPHS * 64) { mx[i] = -__builtin_inff(); sm[i] = 0.0f; }
}

__global__ void k_init_readout(float* readout, float* cnt) {
  int i = blockIdx.x * blockDim.x + threadIdx.x;
  if (i < NUM_GRAPHS * 128) readout[i] = 0.0f;
  if (i < NUM_GRAPHS) cnt[i] = 0.0f;
}

__global__ void k_count(const int* __restrict__ batch, float* cnt, int N) {
  int i = blockIdx.x * blockDim.x + threadIdx.x;
  if (i < N) atomicAdd(cnt + batch[i], 1.0f);
}

__global__ void k_clamp_cnt(float* cnt) {
  int i = threadIdx.x;
  if (i < NUM_GRAPHS) cnt[i] = fmaxf(cnt[i], 1.0f);
}

// ---------------------------------------------------------------------------
// Edge pass 1: e = sum_k att[k]*lrelu(xl[src]+xr[dst]); segment max into m
// ---------------------------------------------------------------------------
__global__ void __launch_bounds__(256)
k_edge_score(const float* __restrict__ xl, const float* __restrict__ xr,
             const float* __restrict__ att, const unsigned char* __restrict__ mask,
             const int* __restrict__ src, const int* __restrict__ dst,
             float* __restrict__ e, float* m, int E) {
  __shared__ float sAtt[64];
  if (threadIdx.x < 64) sAtt[threadIdx.x] = att[threadIdx.x];
  __syncthreads();
  int idx = blockIdx.x * blockDim.x + threadIdx.x;
  if (idx >= E) return;
  int s = src[idx], d = dst[idx];
  const float4* pl = (const float4*)(xl + (size_t)s * 64);
  const float4* pr = (const float4*)(xr + (size_t)d * 64);
  __builtin_prefetch(pl, 0, 1);   // global_prefetch_b8
  float acc = 0.0f;
  #pragma unroll
  for (int q = 0; q < 16; ++q) {
    float4 a = pl[q], b = pr[q];
    float v0 = a.x + b.x, v1 = a.y + b.y, v2 = a.z + b.z, v3 = a.w + b.w;
    v0 = v0 > 0.0f ? v0 : v0 * NEG_SLOPE;
    v1 = v1 > 0.0f ? v1 : v1 * NEG_SLOPE;
    v2 = v2 > 0.0f ? v2 : v2 * NEG_SLOPE;
    v3 = v3 > 0.0f ? v3 : v3 * NEG_SLOPE;
    acc += sAtt[4*q] * v0 + sAtt[4*q+1] * v1 + sAtt[4*q+2] * v2 + sAtt[4*q+3] * v3;
  }
  bool mk = (mask[idx] != 0);
  e[idx] = mk ? acc : -__builtin_inff();
  if (mk) atomicMaxFloat(m + d, acc);
}

// ---------------------------------------------------------------------------
// Edge pass 2: w = mask ? exp(e - finite(m[dst])) : 0; denom[dst] += w
// ---------------------------------------------------------------------------
__global__ void __launch_bounds__(256)
k_edge_weight(const float* __restrict__ e, const float* __restrict__ m,
              const unsigned char* __restrict__ mask, const int* __restrict__ dst,
              float* __restrict__ w, float* den, int E) {
  int idx = blockIdx.x * blockDim.x + threadIdx.x;
  if (idx >= E) return;
  int d = dst[idx];
  float mm = finite_or_zero(m[d]);
  float wv = mask[idx] ? __expf(e[idx] - mm) : 0.0f;
  w[idx] = wv;
  if (wv != 0.0f) atomicAdd(den + d, wv);
}

// ---------------------------------------------------------------------------
// Edge pass 3: agg[dst] += (w/denom[dst]) * xl[src]; 16 threads per edge
// ---------------------------------------------------------------------------
__global__ void __launch_bounds__(256)
k_edge_agg(const float* __restrict__ w, const float* __restrict__ den,
           const float* __restrict__ xl, const int* __restrict__ src,
           const int* __restrict__ dst, float* agg, int E) {
  long long t = (long long)blockIdx.x * blockDim.x + threadIdx.x;
  if (t >= (long long)E * 16) return;
  int eidx = (int)(t >> 4);
  int q = (int)(t & 15);
  float wv = w[eidx];
  if (wv == 0.0f) return;
  int d = dst[eidx], s = src[eidx];
  float alpha = wv / fmaxf(den[d], 1e-16f);
  float4 v = ((const float4*)(xl + (size_t)s * 64))[q];
  float* o = agg + (size_t)d * 64 + q * 4;
  atomicAdd(o + 0, alpha * v.x);
  atomicAdd(o + 1, alpha * v.y);
  atomicAdd(o + 2, alpha * v.z);
  atomicAdd(o + 3, alpha * v.w);
}

// ---------------------------------------------------------------------------
// Node update: h += c*relu(aggH+bH) + (1-c)*relu(aggT+bT); segment max/sum
// ---------------------------------------------------------------------------
__global__ void __launch_bounds__(256)
k_update(float* __restrict__ h, const float* __restrict__ compat,
         const float* __restrict__ aggH, const float* __restrict__ bH,
         const float* __restrict__ aggT, const float* __restrict__ bT,
         const int* __restrict__ batch, float* mx, float* sm, int N) {
  int t = blockIdx.x * blockDim.x + threadIdx.x;
  if (t >= N * 64) return;
  int n = t >> 6, k = t & 63;
  float c  = compat[n];
  float vh = fmaxf(aggH[t] + bH[k], 0.0f);
  float vt = fmaxf(aggT[t] + bT[k], 0.0f);
  float nh = h[t] + c * vh + (1.0f - c) * vt;
  h[t] = nh;
  int g = batch[n];
  atomicMaxFloat(mx + g * 64 + k, nh);
  atomicAdd(sm + g * 64 + k, nh);
}

__global__ void k_readout(const float* __restrict__ mx, const float* __restrict__ sm,
                          const float* __restrict__ cnt, float* readout) {
  int t = blockIdx.x * blockDim.x + threadIdx.x;
  if (t >= NUM_GRAPHS * 64) return;
  int g = t >> 6, k = t & 63;
  readout[g * 128 + k]      += finite_or_zero(mx[t]);
  readout[g * 128 + 64 + k] += sm[t] / cnt[g];
}

// ---------------------------------------------------------------------------
// MLP head + log_softmax: one 128-thread block per graph (trivial cost)
// ---------------------------------------------------------------------------
__global__ void __launch_bounds__(128)
k_head(const float* __restrict__ readout,
       const float* __restrict__ W1, const float* __restrict__ b1,
       const float* __restrict__ W2, const float* __restrict__ b2,
       const float* __restrict__ W3, const float* __restrict__ b3,
       float* __restrict__ out) {
  __shared__ float r[128], z1[128], z2[64], z3[16], red[2];
  int g = blockIdx.x, t = threadIdx.x;
  r[t] = readout[g * 128 + t];
  __syncthreads();
  {
    float a = b1[t];
    for (int k = 0; k < 128; ++k) a += r[k] * W1[k * 128 + t];
    z1[t] = fmaxf(a, 0.0f);
  }
  __syncthreads();
  if (t < 64) {
    float a = b2[t];
    for (int k = 0; k < 128; ++k) a += z1[k] * W2[k * 64 + t];
    z2[t] = fmaxf(a, 0.0f);
  }
  __syncthreads();
  if (t < NUM_CLASSES) {
    float a = b3[t];
    for (int k = 0; k < 64; ++k) a += z2[k] * W3[k * NUM_CLASSES + t];
    z3[t] = a;
  }
  __syncthreads();
  if (t == 0) {
    float mv = z3[0];
    for (int j = 1; j < NUM_CLASSES; ++j) mv = fmaxf(mv, z3[j]);
    float se = 0.0f;
    for (int j = 0; j < NUM_CLASSES; ++j) se += __expf(z3[j] - mv);
    red[0] = mv; red[1] = __logf(se);
  }
  __syncthreads();
  if (t < NUM_CLASSES) out[g * NUM_CLASSES + t] = z3[t] - red[0] - red[1];
}

// ---------------------------------------------------------------------------
// Host-side orchestration
// ---------------------------------------------------------------------------
extern "C" void kernel_launch(void* const* d_in, const int* in_sizes, int n_in,
                              void* d_out, int out_size, void* d_ws, size_t ws_size,
                              hipStream_t stream) {
  const float* x        = (const float*)d_in[0];
  const int*   ei       = (const int*)  d_in[1];
  const int*   batch    = (const int*)  d_in[2];
  const unsigned char* hom_mask = (const unsigned char*)d_in[3];
  const unsigned char* het_mask = (const unsigned char*)d_in[4];
  const float* compat   = (const float*)d_in[5];
  const float* pre_W    = (const float*)d_in[6];
  const float* pre_b    = (const float*)d_in[7];
  const float* hom_Wl   = (const float*)d_in[8];
  const float* hom_Wr   = (const float*)d_in[9];
  const float* hom_att  = (const float*)d_in[10];
  const float* hom_b    = (const float*)d_in[11];
  const float* het_Wl   = (const float*)d_in[12];
  const float* het_Wr   = (const float*)d_in[13];
  const float* het_att  = (const float*)d_in[14];
  const float* het_b    = (const float*)d_in[15];
  const float* lin1_W   = (const float*)d_in[16];
  const float* lin1_b   = (const float*)d_in[17];
  const float* lin2_W   = (const float*)d_in[18];
  const float* lin2_b   = (const float*)d_in[19];
  const float* lin3_W   = (const float*)d_in[20];
  const float* lin3_b   = (const float*)d_in[21];

  const int N = N_NODES;
  const int E = in_sizes[1] / 2;
  const int* src = ei;
  const int* dst = ei + E;

  // workspace layout (floats)
  float* base = (float*)d_ws;
  size_t off = 0;
  auto alloc = [&](size_t n) { float* p = base + off; off += n; return p; };
  float* h       = alloc((size_t)N * 64);
  float* xl      = alloc((size_t)N * 64);
  float* xr      = alloc((size_t)N * 64);
  float* aggH    = alloc((size_t)N * 64);
  float* aggT    = alloc((size_t)N * 64);
  float* ebuf    = alloc((size_t)E);
  float* wbuf    = alloc((size_t)E);
  float* mbuf    = alloc((size_t)N);
  float* den     = alloc((size_t)N);
  float* mx      = alloc((size_t)NUM_GRAPHS * 64);
  float* sm      = alloc((size_t)NUM_GRAPHS * 64);
  float* readout = alloc((size_t)NUM_GRAPHS * 128);
  float* cntf    = alloc((size_t)NUM_GRAPHS);
  (void)ws_size; (void)n_in; (void)out_size;

  const int gNF  = (N * 64 + 255) / 256;        // node-feature grid
  const int gE   = (E + 255) / 256;             // edge grid
  const int gE16 = (int)(((long long)E * 16 + 255) / 256);
  const int gGemm = (N + 127) / 128;

  // counts + readout init
  k_init_readout<<<(NUM_GRAPHS * 128 + 255) / 256, 256, 0, stream>>>(readout, cntf);
  k_count<<<(N + 255) / 256, 256, 0, stream>>>(batch, cntf, N);
  k_clamp_cnt<<<1, 256, 0, stream>>>(cntf);

  // pre-linear: h = x @ pre_W + pre_b  (WMMA, K=128)
  k_gemm64<NUM_FEAT><<<gGemm, 256, 0, stream>>>(x, pre_W, pre_b, h, N);

  auto conv = [&](const float* Wl, const float* Wr, const float* att,
                  const unsigned char* mask, float* agg) {
    k_gemm64<NHID><<<gGemm, 256, 0, stream>>>(h, Wl, nullptr, xl, N);
    k_gemm64<NHID><<<gGemm, 256, 0, stream>>>(h, Wr, nullptr, xr, N);
    k_init_conv<<<gNF, 256, 0, stream>>>(mbuf, den, agg, N);
    k_edge_score<<<gE, 256, 0, stream>>>(xl, xr, att, mask, src, dst, ebuf, mbuf, E);
    k_edge_weight<<<gE, 256, 0, stream>>>(ebuf, mbuf, mask, dst, wbuf, den, E);
    k_edge_agg<<<gE16, 256, 0, stream>>>(wbuf, den, xl, src, dst, agg, E);
  };

  for (int layer = 0; layer < 3; ++layer) {
    const size_t wOff = (size_t)layer * NHID * NHID;
    const size_t vOff = (size_t)layer * NHID;
    conv(hom_Wl + wOff, hom_Wr + wOff, hom_att + vOff, hom_mask, aggH);
    conv(het_Wl + wOff, het_Wr + wOff, het_att + vOff, het_mask, aggT);
    k_init_graphseg<<<(NUM_GRAPHS * 64 + 255) / 256, 256, 0, stream>>>(mx, sm);
    k_update<<<gNF, 256, 0, stream>>>(h, compat, aggH, hom_b + vOff,
                                      aggT, het_b + vOff, batch, mx, sm, N);
    k_readout<<<(NUM_GRAPHS * 64 + 255) / 256, 256, 0, stream>>>(mx, sm, cntf, readout);
  }

  k_head<<<NUM_GRAPHS, 128, 0, stream>>>(readout, lin1_W, lin1_b, lin2_W, lin2_b,
                                         lin3_W, lin3_b, (float*)d_out);
}
